// SelfAttention_26774826123280
// MI455X (gfx1250) — matrix operand order repaired and checked by reading, not compiled, not run on hardware
//
#include <hip/hip_runtime.h>
#include <hip/hip_bf16.h>
#include <stdint.h>

// ---------------------------------------------------------------------------
// CDNA5 (gfx1250, wave32) self-attention block: RMSNorm -> QKV -> flash attn
// -> proj + residual.  All GEMM math through v_wmma_f32_16x16x32_bf16 with a
// 2-stage ping-pong pipeline (explicit unroll-by-2: no register rotation
// copies, WMMAs overlap in-flight loads via partial s_wait_loadcnt).
// ---------------------------------------------------------------------------

typedef __attribute__((ext_vector_type(16))) __bf16 v16bf;
typedef __attribute__((ext_vector_type(8)))  float  v8f;

#define D_MODEL 1024
#define NHEADS  16
#define HDIM    64
#define LSEQ    2048
#define BATCH   2
#define SROWS   (BATCH * LSEQ)   // 4096 token rows

// float -> bf16 (round to nearest even), returned as raw u16
__device__ __forceinline__ unsigned short f2bf_u16(float f) {
    union { float f; uint32_t u; } v; v.f = f;
    uint32_t r = v.u + 0x7FFFu + ((v.u >> 16) & 1u);
    return (unsigned short)(r >> 16);
}

union BF16Frag {
    v16bf v;
    unsigned short u[16];
    uint4 q[2];
};

// A-fragment (16x32 bf16) from a row-major bf16 matrix.
// ISA layout: lane<16 -> row=lane, K in {0..7} u {16..23};
//             lane>=16 -> row=lane-16, K in {8..15} u {24..31}.
__device__ __forceinline__ v16bf load_a_frag(const unsigned short* src, int ld,
                                             int row0, int k0, int lane) {
    int r    = row0 + (lane & 15);
    int koff = (lane < 16) ? 0 : 8;
    const unsigned short* p = src + (size_t)r * ld + k0 + koff;
    BF16Frag f;
    f.q[0] = *reinterpret_cast<const uint4*>(p);        // K = k0+koff .. +7
    f.q[1] = *reinterpret_cast<const uint4*>(p + 16);   // K = k0+16+koff .. +7
    return f.v;
}

// B-fragment (32x16 bf16) from "transposed" storage: srcT[n][k] contiguous in k.
// ISA layout: lane<16 -> col=lane, K in {0..7} u {16..23};
//             lane>=16 -> col=lane-16, K in {8..15} u {24..31}.
__device__ __forceinline__ v16bf load_bT_frag(const unsigned short* srcT, int ld,
                                              int n0, int k0, int lane) {
    int n  = n0 + (lane & 15);
    int kb = (lane < 16) ? 0 : 8;
    const unsigned short* p = srcT + (size_t)n * ld + k0 + kb;
    BF16Frag f;
    f.q[0] = *reinterpret_cast<const uint4*>(p);
    f.q[1] = *reinterpret_cast<const uint4*>(p + 16);
    return f.v;
}

__device__ __forceinline__ v8f wmma_bf16(v16bf a, v16bf b, v8f c) {
    return __builtin_amdgcn_wmma_f32_16x16x32_bf16(
        /*neg_a=*/false, a, /*neg_b=*/false, b,
        /*c_mod=*/(short)0, c, /*reuse_a=*/false, /*reuse_b=*/false);
}

// ---------------------------------------------------------------------------
// Kernel 1: RMSNorm (f32 in) -> bf16 h
// ---------------------------------------------------------------------------
__global__ void __launch_bounds__(256)
rmsnorm_kernel(const float* __restrict__ x, const float* __restrict__ w,
               unsigned short* __restrict__ h) {
    const int row = blockIdx.x;                 // 0..SROWS-1
    const float* xr = x + (size_t)row * D_MODEL;
    __shared__ float red[256];
    float ss = 0.f;
    for (int i = threadIdx.x; i < D_MODEL; i += 256) {
        float v = xr[i];
        ss += v * v;
    }
    red[threadIdx.x] = ss;
    __syncthreads();
    for (int s = 128; s > 0; s >>= 1) {
        if (threadIdx.x < (unsigned)s) red[threadIdx.x] += red[threadIdx.x + s];
        __syncthreads();
    }
    const float inv = rsqrtf(red[0] * (1.0f / D_MODEL) + 1e-6f);
    for (int i = threadIdx.x; i < D_MODEL; i += 256)
        h[(size_t)row * D_MODEL + i] = f2bf_u16(xr[i] * inv * w[i]);
}

// ---------------------------------------------------------------------------
// Kernel 2: f32 weight [K,N] -> bf16 transposed [N,K]
// ---------------------------------------------------------------------------
__global__ void __launch_bounds__(256)
convert_transpose_kernel(const float* __restrict__ w, unsigned short* __restrict__ wT,
                         int K, int N) {
    int idx = blockIdx.x * 256 + threadIdx.x;
    if (idx >= K * N) return;
    int k = idx / N, n = idx % N;               // coalesced read over n
    wT[(size_t)n * K + k] = f2bf_u16(w[idx]);
}

// ---------------------------------------------------------------------------
// GEMM core: 16x64 wave tile, K loop stepping 64 with A/B ping-pong stages.
// ---------------------------------------------------------------------------
__device__ __forceinline__ void gemm_16x64_pipelined(
    const unsigned short* __restrict__ amat, const unsigned short* __restrict__ wT,
    int m0, int n0, int lane, v8f (&acc)[4]) {
    v16bf aA, bA[4], aB, bB[4];

    aA = load_a_frag(amat, D_MODEL, m0, 0, lane);
#pragma unroll
    for (int c = 0; c < 4; ++c)
        bA[c] = load_bT_frag(wT, D_MODEL, n0 + c * 16, 0, lane);

    for (int kk = 0; kk < D_MODEL; kk += 64) {
        // stage B loads (kk+32) -- kk+32 <= 992, always in range
        aB = load_a_frag(amat, D_MODEL, m0, kk + 32, lane);
#pragma unroll
        for (int c = 0; c < 4; ++c)
            bB[c] = load_bT_frag(wT, D_MODEL, n0 + c * 16, kk + 32, lane);
        __builtin_prefetch(amat + (size_t)(m0 + (lane & 15)) * D_MODEL + kk + 128, 0, 1);

        // stage A compute (kk)
#pragma unroll
        for (int c = 0; c < 4; ++c)
            acc[c] = wmma_bf16(aA, bA[c], acc[c]);

        // stage A loads (kk+64, clamped on last iteration)
        const int kn2 = (kk + 64 < D_MODEL) ? kk + 64 : 0;
        aA = load_a_frag(amat, D_MODEL, m0, kn2, lane);
#pragma unroll
        for (int c = 0; c < 4; ++c)
            bA[c] = load_bT_frag(wT, D_MODEL, n0 + c * 16, kn2, lane);

        // stage B compute (kk+32)
#pragma unroll
        for (int c = 0; c < 4; ++c)
            acc[c] = wmma_bf16(aB, bB[c], acc[c]);
    }
}

// ---------------------------------------------------------------------------
// Kernel 3: QKV GEMM  h[S,D] @ w_qkv -> Q[b,h,l,hd]*scale, K[b,h,l,hd], V^T[b,h,hd,l]
// 8 waves/block, wave tile 16x64, block tile 64x128.
// ---------------------------------------------------------------------------
__global__ void __launch_bounds__(256)
qkv_gemm_kernel(const unsigned short* __restrict__ h,
                const unsigned short* __restrict__ wT,   // [3D, D] bf16
                unsigned short* __restrict__ qb,
                unsigned short* __restrict__ kb,
                unsigned short* __restrict__ vT) {
    const int lane = threadIdx.x & 31;
    const int wid  = threadIdx.x >> 5;
    const int wr   = wid >> 1;                  // 0..3
    const int wc   = wid & 1;                   // 0..1
    const int m0 = blockIdx.y * 64  + wr * 16;
    const int n0 = blockIdx.x * 128 + wc * 64;

    v8f acc[4] = {};
    gemm_16x64_pipelined(h, wT, m0, n0, lane, acc);

    // Scatter C fragments (lane<16: col=lane rows 0..7 ; lane>=16: rows 8..15)
    const int col = lane & 15;
    const int rb  = (lane < 16) ? 0 : 8;
#pragma unroll
    for (int c = 0; c < 4; ++c) {
        const int nl    = n0 + c * 16;          // multiple of 16; q/k/v + head uniform
        const int which = nl >> 10;             // 0=Q 1=K 2=V
        const int nn    = nl & 1023;
        const int head  = nn >> 6;
        const int hd    = (nn & 63) + col;
#pragma unroll
        for (int r = 0; r < 8; ++r) {
            const int m  = m0 + rb + r;
            const int bb = m >> 11;             // / LSEQ
            const int l  = m & (LSEQ - 1);
            const float val = acc[c][r];
            const size_t bh = (size_t)(bb * NHEADS + head);
            if (which == 0)
                qb[(bh * LSEQ + l) * HDIM + hd] = f2bf_u16(val * 0.125f); // fold 1/sqrt(64)
            else if (which == 1)
                kb[(bh * LSEQ + l) * HDIM + hd] = f2bf_u16(val);
            else
                vT[(bh * HDIM + hd) * LSEQ + l] = f2bf_u16(val);          // transposed
        }
    }
}

// ---------------------------------------------------------------------------
// Flash-attention step: one 32-key block.  Consumes K fragments `kf`, emits
// K fragments `kn` for the next block (loaded early so the softmax VALU/exp
// chain overlaps them).  Force-inlined; called twice with A/B buffers swapped
// so there is no register rotation.
// ---------------------------------------------------------------------------
__device__ __forceinline__ void attn_step(
    const unsigned short* __restrict__ K, const unsigned short* __restrict__ VT,
    v16bf qf0, v16bf qf1,
    v16bf (&kf)[4], v16bf (&kn)[4], int kblk, int nb,
    float (&mrow)[8], float (&lrow)[8], v8f (&acc)[4],
    unsigned short (&ptile)[16][32], int lane) {
    // ---- scores: S = Q @ K^T  (two 16x16 tiles over 32 keys) ----
    v8f s0 = {}, s1 = {};
    s0 = wmma_bf16(qf0, kf[0], s0);
    s0 = wmma_bf16(qf1, kf[1], s0);
    s1 = wmma_bf16(qf0, kf[2], s1);
    s1 = wmma_bf16(qf1, kf[3], s1);

    // ---- issue V loads (this block) + K loads (next block) now ----
    v16bf vf[4];
#pragma unroll
    for (int c = 0; c < 4; ++c)
        vf[c] = load_bT_frag(VT, LSEQ, c * 16, kblk, lane);
    kn[0] = load_bT_frag(K, HDIM, nb,      0,  lane);
    kn[1] = load_bT_frag(K, HDIM, nb,      32, lane);
    kn[2] = load_bT_frag(K, HDIM, nb + 16, 0,  lane);
    kn[3] = load_bT_frag(K, HDIM, nb + 16, 32, lane);

    // ---- row max over this key block (reduce across the 16-lane half) ----
    float mt[8];
#pragma unroll
    for (int r = 0; r < 8; ++r) mt[r] = fmaxf(s0[r], s1[r]);
#pragma unroll
    for (int off = 1; off < 16; off <<= 1) {
#pragma unroll
        for (int r = 0; r < 8; ++r)
            mt[r] = fmaxf(mt[r], __shfl_xor(mt[r], off, 32));
    }

    // ---- online softmax update ----
    float p0[8], p1[8], rs[8];
#pragma unroll
    for (int r = 0; r < 8; ++r) {
        const float mnew = fmaxf(mrow[r], mt[r]);
        const float cf   = __expf(mrow[r] - mnew);
        mrow[r] = mnew;
        p0[r] = __expf(s0[r] - mnew);
        p1[r] = __expf(s1[r] - mnew);
        rs[r] = p0[r] + p1[r];
        lrow[r] *= cf;
#pragma unroll
        for (int c = 0; c < 4; ++c) acc[c][r] *= cf;
    }
#pragma unroll
    for (int off = 1; off < 16; off <<= 1) {
#pragma unroll
        for (int r = 0; r < 8; ++r) rs[r] += __shfl_xor(rs[r], off, 32);
    }
#pragma unroll
    for (int r = 0; r < 8; ++r) lrow[r] += rs[r];

    // ---- P (C-layout, lane=col) -> LDS -> A-fragment (lane=row) ----
    // The staging tile is wave-private and DS ops from one wave execute in
    // order, so a dscnt drain (not a workgroup barrier) is sufficient.
    const int col = lane & 15;
    const int rb  = (lane < 16) ? 0 : 8;
#pragma unroll
    for (int r = 0; r < 8; ++r) {
        ptile[rb + r][col]      = f2bf_u16(p0[r]);
        ptile[rb + r][16 + col] = f2bf_u16(p1[r]);
    }
    asm volatile("s_wait_dscnt 0x0" ::: "memory");
    const v16bf pf = load_a_frag(&ptile[0][0], 32, 0, 0, lane);

    // ---- ctx += P @ V  (V^T rows are contiguous in key index) ----
#pragma unroll
    for (int c = 0; c < 4; ++c)
        acc[c] = wmma_bf16(pf, vf[c], acc[c]);
}

// ---------------------------------------------------------------------------
// Kernel 4: flash attention.  4 waves/block; wave owns 16 query rows; keys
// streamed in 32-key blocks, two blocks per iteration (ping-pong K frags).
// ---------------------------------------------------------------------------
__global__ void __launch_bounds__(128)
attn_kernel(const unsigned short* __restrict__ q,
            const unsigned short* __restrict__ k,
            const unsigned short* __restrict__ vT,
            unsigned short* __restrict__ ctx) {
    const int lane  = threadIdx.x & 31;
    const int wid   = threadIdx.x >> 5;
    const int bh    = blockIdx.y;               // b*NHEADS + head
    const int b     = bh >> 4;
    const int head  = bh & 15;
    const int qbase = blockIdx.x * 64 + wid * 16;

    const unsigned short* Q  = q  + (size_t)bh * LSEQ * HDIM;
    const unsigned short* K  = k  + (size_t)bh * LSEQ * HDIM;
    const unsigned short* VT = vT + (size_t)bh * HDIM * LSEQ;

    const v16bf qf0 = load_a_frag(Q, HDIM, qbase, 0,  lane);
    const v16bf qf1 = load_a_frag(Q, HDIM, qbase, 32, lane);

    float mrow[8], lrow[8];
    v8f acc[4] = {};
#pragma unroll
    for (int r = 0; r < 8; ++r) { mrow[r] = -1e30f; lrow[r] = 0.f; }

    __shared__ __align__(16) unsigned short plds[4][16][32];  // 1 KB per wave

    // prologue: K fragments for key block 0 into stage A
    v16bf kfA[4], kfB[4];
    kfA[0] = load_bT_frag(K, HDIM, 0,  0,  lane);
    kfA[1] = load_bT_frag(K, HDIM, 0,  32, lane);
    kfA[2] = load_bT_frag(K, HDIM, 16, 0,  lane);
    kfA[3] = load_bT_frag(K, HDIM, 16, 32, lane);

    for (int kblk = 0; kblk < LSEQ; kblk += 64) {
        attn_step(K, VT, qf0, qf1, kfA, kfB, kblk, kblk + 32,
                  mrow, lrow, acc, plds[wid], lane);
        const int nb2 = (kblk + 64 < LSEQ) ? kblk + 64 : 0;   // clamp: in bounds
        attn_step(K, VT, qf0, qf1, kfB, kfA, kblk + 32, nb2,
                  mrow, lrow, acc, plds[wid], lane);
    }

    // ---- normalize and store ctx as bf16 [b, l, head*64+hd] ----
    const int col = lane & 15;
    const int rb  = (lane < 16) ? 0 : 8;
#pragma unroll
    for (int c = 0; c < 4; ++c) {
#pragma unroll
        for (int r = 0; r < 8; ++r) {
            const int qrow = qbase + rb + r;
            const float o  = acc[c][r] / lrow[r];
            ctx[((size_t)(b * LSEQ + qrow)) * D_MODEL + head * HDIM + c * 16 + col] =
                f2bf_u16(o);
        }
    }
}

// ---------------------------------------------------------------------------
// Kernel 5: proj GEMM ctx[S,D] @ w_proj + residual -> out f32
// ---------------------------------------------------------------------------
__global__ void __launch_bounds__(256)
proj_gemm_kernel(const unsigned short* __restrict__ ctx,
                 const unsigned short* __restrict__ wT,   // [D, D] bf16
                 const float* __restrict__ x,
                 float* __restrict__ out) {
    const int lane = threadIdx.x & 31;
    const int wid  = threadIdx.x >> 5;
    const int wr   = wid >> 1;
    const int wc   = wid & 1;
    const int m0 = blockIdx.y * 64  + wr * 16;
    const int n0 = blockIdx.x * 128 + wc * 64;

    v8f acc[4] = {};
    gemm_16x64_pipelined(ctx, wT, m0, n0, lane, acc);

    const int col = lane & 15;
    const int rb  = (lane < 16) ? 0 : 8;
#pragma unroll
    for (int c = 0; c < 4; ++c) {
        const int n = n0 + c * 16 + col;
#pragma unroll
        for (int r = 0; r < 8; ++r) {
            const int m = m0 + rb + r;
            const size_t idx = (size_t)m * D_MODEL + n;
            out[idx] = x[idx] + acc[c][r];      // residual in f32
        }
    }
}

// ---------------------------------------------------------------------------
// Host launcher
// ---------------------------------------------------------------------------
extern "C" void kernel_launch(void* const* d_in, const int* in_sizes, int n_in,
                              void* d_out, int out_size, void* d_ws, size_t ws_size,
                              hipStream_t stream) {
    const float* x      = (const float*)d_in[0];   // [2,2048,1024]
    const float* norm_w = (const float*)d_in[1];   // [1024]
    const float* w_qkv  = (const float*)d_in[2];   // [1024,3072]
    const float* w_proj = (const float*)d_in[3];   // [1024,1024]
    float* out = (float*)d_out;

    // Workspace carve-up (bf16 buffers), ~48 MiB total.
    uint8_t* p = (uint8_t*)d_ws;
    unsigned short* h      = (unsigned short*)p; p += (size_t)SROWS * D_MODEL * 2;     // 8 MiB
    unsigned short* wqkvT  = (unsigned short*)p; p += (size_t)3 * D_MODEL * D_MODEL * 2; // 6 MiB
    unsigned short* wprojT = (unsigned short*)p; p += (size_t)D_MODEL * D_MODEL * 2;   // 2 MiB
    unsigned short* qb     = (unsigned short*)p; p += (size_t)SROWS * D_MODEL * 2;     // 8 MiB
    unsigned short* kbuf   = (unsigned short*)p; p += (size_t)SROWS * D_MODEL * 2;     // 8 MiB
    unsigned short* vT     = (unsigned short*)p; p += (size_t)SROWS * D_MODEL * 2;     // 8 MiB
    unsigned short* ctxb   = (unsigned short*)p; p += (size_t)SROWS * D_MODEL * 2;     // 8 MiB
    (void)ws_size; (void)in_sizes; (void)n_in; (void)out_size;

    rmsnorm_kernel<<<SROWS, 256, 0, stream>>>(x, norm_w, h);

    {
        int total = D_MODEL * 3 * D_MODEL;
        convert_transpose_kernel<<<(total + 255) / 256, 256, 0, stream>>>(
            w_qkv, wqkvT, D_MODEL, 3 * D_MODEL);
    }
    {
        int total = D_MODEL * D_MODEL;
        convert_transpose_kernel<<<(total + 255) / 256, 256, 0, stream>>>(
            w_proj, wprojT, D_MODEL, D_MODEL);
    }

    // QKV: block tile 64x128 over [4096, 3072]
    qkv_gemm_kernel<<<dim3(3 * D_MODEL / 128, SROWS / 64), 256, 0, stream>>>(
        h, wqkvT, qb, kbuf, vT);

    // Attention: grid (query tiles of 64, b*h)
    attn_kernel<<<dim3(LSEQ / 64, BATCH * NHEADS), 128, 0, stream>>>(
        qb, kbuf, vT, ctxb);

    // Proj + residual: block tile 64x128 over [4096, 1024]
    proj_gemm_kernel<<<dim3(D_MODEL / 128, SROWS / 64), 256, 0, stream>>>(
        ctxb, wprojT, x, out);
}
